// SelfAttn_29454885716718
// MI455X (gfx1250) — matrix operand order repaired and checked by reading, not compiled, not run on hardware
//
#include <hip/hip_runtime.h>
#include <hip/hip_bf16.h>

// ---------------- types ----------------
typedef __bf16 v8bf  __attribute__((ext_vector_type(8)));
typedef __bf16 v16bf __attribute__((ext_vector_type(16)));
typedef float  v8f   __attribute__((ext_vector_type(8)));
typedef unsigned v4u __attribute__((ext_vector_type(4)));
typedef int      v4i __attribute__((ext_vector_type(4)));
typedef int      v8i __attribute__((ext_vector_type(8)));

#define DMODEL 1024
#define NSEQ   2048
#define NBATCH 4
#define MTOT   (NBATCH * NSEQ)   // 8192

static __device__ __forceinline__ v8f zero8() {
    v8f v;
#pragma unroll
    for (int i = 0; i < 8; ++i) v[i] = 0.0f;
    return v;
}

// LDS byte offset of a generic pointer that points into __shared__ memory.
static __device__ __forceinline__ unsigned lds_off_u32(const void* p) {
    return (unsigned)(unsigned long long)
        (__attribute__((address_space(3))) const char*)(const char*)p;
}

// A-fragment (16x32 bf16) from bf16 source: lane holds K = klo..+7, klo+16..+23.
static __device__ __forceinline__ v16bf load_a16(const __bf16* p) {
    v8bf lo = *(const v8bf*)(p);
    v8bf hi = *(const v8bf*)(p + 16);
    v16bf r;
#pragma unroll
    for (int i = 0; i < 8; ++i) { r[i] = lo[i]; r[i + 8] = hi[i]; }
    return r;
}

// A-fragment from an fp32 tile in LDS (TDM-loaded), converting to bf16.
static __device__ __forceinline__ v16bf load_a16_f32(const float* p) {
    v16bf r;
#pragma unroll
    for (int i = 0; i < 8; ++i) {
        r[i]     = (__bf16)p[i];
        r[i + 8] = (__bf16)p[i + 16];
    }
    return r;
}

// B-fragment (32x16 bf16): lane holds one column, K = klo..+7, klo+16..+23.
static __device__ __forceinline__ v16bf load_b16(const __bf16* p) {
    v8bf lo = *(const v8bf*)(p);
    v8bf hi = *(const v8bf*)(p + 16);
    v16bf r;
#pragma unroll
    for (int i = 0; i < 8; ++i) { r[i] = lo[i]; r[i + 8] = hi[i]; }
    return r;
}

static __device__ __forceinline__ v8f wmma_bf16(v16bf a, v16bf b, v8f c) {
    return __builtin_amdgcn_wmma_f32_16x16x32_bf16(false, a, false, b, (short)0, c,
                                                   false, false);
}

// TDM: async-load a 2D fp32 tile [rows=128 x cols=32], row stride DMODEL,
// from `gaddr` into LDS at byte offset `ldsoff`.  Issued per-wave (EXEC ignored).
static __device__ __forceinline__ void tdm_load_a_tile(unsigned ldsoff,
                                                       const float* gaddr) {
    unsigned long long ga = (unsigned long long)gaddr;
    v4u g0;
    g0[0] = 1u;                                    // count=1, user descriptor
    g0[1] = ldsoff;                                // lds_addr
    g0[2] = (unsigned)ga;                          // global_addr[31:0]
    g0[3] = (unsigned)(ga >> 32) | (2u << 30);     // global_addr[56:32] | type=2
    v8i g1;
    g1[0] = 2 << 16;                               // data_size = 4 bytes
    g1[1] = DMODEL << 16;                          // tensor_dim0[15:0]
    g1[2] = (int)((unsigned)MTOT << 16);           // tensor_dim1[15:0]
    g1[3] = 32 << 16;                              // tile_dim0 = 32
    g1[4] = 128;                                   // tile_dim1 = 128
    g1[5] = DMODEL;                                // tensor_dim0_stride
    g1[6] = 0;
    g1[7] = 0;
    v4i z4 = {0, 0, 0, 0};
    v8i z8 = {0, 0, 0, 0, 0, 0, 0, 0};
    __builtin_amdgcn_tensor_load_to_lds(g0, g1, z4, z4, z8, 0);
}

// ---------------------------------------------------------------------------
// Phase 1: O = x @ W  (fp32 in, bf16 out).  128x128 block tile, 8 waves.
// TRANS==0: out[m][n] row-major [MTOT][DMODEL]   (Q, K)
// TRANS==1: out = V^T layout: out[b][d][n]       ([NBATCH][DMODEL][NSEQ])
// A tile double-buffered through the Tensor Data Mover.
// ---------------------------------------------------------------------------
template <int TRANS>
__global__ __launch_bounds__(256) void qkv_gemm(const float* __restrict__ x,
                                                const float* __restrict__ W,
                                                __bf16* __restrict__ out) {
    __shared__ char smem_raw[2 * 16384 + 8192];  // 40KB
    float*  sA0 = (float*)smem_raw;              // [128][32] fp32 TDM buf 0
    float*  sA1 = (float*)(smem_raw + 16384);    // [128][32] fp32 TDM buf 1
    __bf16* sB  = (__bf16*)(smem_raw + 32768);   // [n=128][k=32] bf16   8KB
    __bf16* sC  = (__bf16*)smem_raw;             // epilogue: [128][128] bf16 32KB

    const int tid  = threadIdx.x;
    const int lane = tid & 31;
    const int wave = __builtin_amdgcn_readfirstlane(tid >> 5);
    const int wm   = (wave >> 2) * 64;
    const int wn   = (wave & 3) * 32;
    const int m0   = blockIdx.x * 128;
    const int n0   = blockIdx.y * 128;
    const int l15  = lane & 15;
    const int hi8  = (lane >= 16) ? 8 : 0;
    const int klo  = (lane >= 16) ? 8 : 0;
    const int kb   = (lane >= 16) ? 16 : 0;
    const unsigned ldsA[2] = {lds_off_u32(sA0), lds_off_u32(sA1)};
    float* sAbuf[2] = {sA0, sA1};

    v8f acc[4][2];
#pragma unroll
    for (int i = 0; i < 4; ++i)
#pragma unroll
        for (int j = 0; j < 2; ++j) acc[i][j] = zero8();

    // prime the TDM pipeline
    if (wave == 0)
        tdm_load_a_tile(ldsA[0], x + (size_t)m0 * DMODEL);

    const int NSTEP = DMODEL / 32;
    for (int i = 0; i < NSTEP; ++i) {
        const int k0 = i * 32;
        __syncthreads();  // prior readers of buf[(i+1)&1] are done
        if (wave == 0 && i + 1 < NSTEP)
            tdm_load_a_tile(ldsA[(i + 1) & 1],
                            x + (size_t)m0 * DMODEL + k0 + 32);

        // stage B tile 32x128 fp32 -> bf16, transposed into [n][k]
#pragma unroll
        for (int j = 0; j < 4; ++j) {
            int lin = tid + j * 256;
            int k = lin >> 5, q = (lin & 31) * 4;
            const float4 v = *(const float4*)(W + (size_t)(k0 + k) * DMODEL + n0 + q);
            sB[(q + 0) * 32 + k] = (__bf16)v.x;
            sB[(q + 1) * 32 + k] = (__bf16)v.y;
            sB[(q + 2) * 32 + k] = (__bf16)v.z;
            sB[(q + 3) * 32 + k] = (__bf16)v.w;
        }
        // prefetch next W tile into L2
        if (k0 + 32 < DMODEL)
            __builtin_prefetch(W + (size_t)(k0 + 32 + (tid >> 3)) * DMODEL + n0 +
                                   (tid & 7) * 16, 0, 3);

        // wait for the *current* tile only (TDM ops of a wave finish in order)
        if (wave == 0) {
            if (i + 1 < NSTEP)
                __builtin_amdgcn_s_wait_tensorcnt(1);
            else
                __builtin_amdgcn_s_wait_tensorcnt(0);
        }
        __syncthreads();

        const float* sAf = sAbuf[i & 1];
        v16bf bfrag[2];
#pragma unroll
        for (int tn = 0; tn < 2; ++tn)
            bfrag[tn] = load_b16(&sB[(wn + tn * 16 + l15) * 32 + klo]);
#pragma unroll
        for (int tm = 0; tm < 4; ++tm) {
            v16bf a = load_a16_f32(&sAf[(wm + tm * 16 + l15) * 32 + klo]);
#pragma unroll
            for (int tn = 0; tn < 2; ++tn)
                acc[tm][tn] = wmma_bf16(a, bfrag[tn], acc[tm][tn]);
        }
    }

    if constexpr (TRANS) {
        // V^T: lane's 8 accumulator rows are contiguous n positions -> b128 store
#pragma unroll
        for (int tm = 0; tm < 4; ++tm)
#pragma unroll
            for (int tn = 0; tn < 2; ++tn) {
                int col  = n0 + wn + tn * 16 + l15;          // d
                int row0 = m0 + wm + tm * 16 + hi8;          // first of 8 rows
                int bi = row0 >> 11, nb = row0 & (NSEQ - 1);
                v8bf pk;
#pragma unroll
                for (int r = 0; r < 8; ++r) pk[r] = (__bf16)acc[tm][tn][r];
                *(v8bf*)(out + ((size_t)bi * DMODEL + col) * NSEQ + nb) = pk;
            }
    } else {
        // Q/K: reorganize through LDS, then fully coalesced b128 stores
        __syncthreads();
#pragma unroll
        for (int tm = 0; tm < 4; ++tm)
#pragma unroll
            for (int tn = 0; tn < 2; ++tn) {
                int cl = wn + tn * 16 + l15;
#pragma unroll
                for (int r = 0; r < 8; ++r)
                    sC[(wm + tm * 16 + r + hi8) * 128 + cl] = (__bf16)acc[tm][tn][r];
            }
        __syncthreads();
#pragma unroll
        for (int j = 0; j < 8; ++j) {
            int idx = tid + j * 256;           // 2048 uint4 total
            int row = idx >> 4, q = idx & 15;  // 16 uint4 per 128-col row
            *(uint4*)(out + (size_t)(m0 + row) * DMODEL + n0 + q * 8) =
                ((const uint4*)sC)[idx];
        }
    }
}

// ---------------------------------------------------------------------------
// Scores inner loop for NG live 4-tile groups: straight-line, with an explicit
// one-tile-deep load rotation so each WMMA has the next tile's loads in flight.
// ---------------------------------------------------------------------------
template <int NG>
static __device__ __forceinline__ void score_loop(v8f (&sacc)[16],
                                                  const __bf16* __restrict__ qrow,
                                                  const __bf16* __restrict__ kbase,
                                                  int klo, int kb) {
    constexpr int NT = NG * 4;
    for (int d0 = 0; d0 < DMODEL; d0 += 32) {
        v16bf a  = load_a16(qrow + d0 + klo);
        v16bf nb = load_b16(kbase + d0 + klo);
#pragma unroll
        for (int t = 0; t < NT; ++t) {
            v16bf cb = nb;
            if (t + 1 < NT)
                nb = load_b16(kbase + (size_t)((t + 1) * 16) * DMODEL + d0 + klo);
            sacc[t] = wmma_bf16(a, cb, sacc[t]);
        }
    }
}

// ---------------------------------------------------------------------------
// Phase 2: causal attention, one block per 16-query tile, 8 waves.
// ---------------------------------------------------------------------------
__global__ __launch_bounds__(256) void attn_kernel(const __bf16* __restrict__ Qb,
                                                   const __bf16* __restrict__ Kb,
                                                   const __bf16* __restrict__ Vt,
                                                   float* __restrict__ out,
                                                   float scale) {
    extern __shared__ char smem[];
    __bf16* sP   = (__bf16*)smem;                   // [16][NSEQ] bf16 = 64KB
    float*  sO   = (float*)smem;                    // epilogue: [16][DMODEL] f32 64KB
    float*  red  = (float*)(smem + 16 * NSEQ * 2);  // [8][16] partials
    float*  rsta = red + 8 * 16;                    // [0..15]=rowmax [16..31]=1/sum

    const int tid  = threadIdx.x;
    const int lane = tid & 31;
    const int wave = __builtin_amdgcn_readfirstlane(tid >> 5);  // scalar wave id
    const int l15  = lane & 15;
    const int hi8  = (lane >= 16) ? 8 : 0;
    const int klo  = (lane >= 16) ? 8 : 0;
    const int kb   = (lane >= 16) ? 16 : 0;
    const int q0   = blockIdx.x * 16;
    const int b    = blockIdx.y;
    const int qlast = q0 + 15;

    // zero P (causally-dead tiles must read as 0 in the PV phase)
    for (int i = tid; i < (16 * NSEQ) / 8; i += 256)
        ((uint4*)sP)[i] = make_uint4(0u, 0u, 0u, 0u);
    __syncthreads();

    // ---------------- scores: S = Q K^T * scale, causal ----------------
    const int nwb = wave * 256;  // scalar
    v8f sacc[16];
#pragma unroll
    for (int t = 0; t < 16; ++t) sacc[t] = zero8();

    const __bf16* qrow = Qb + ((size_t)b * NSEQ + q0 + l15) * DMODEL;
    const __bf16* kbase = Kb + ((size_t)b * NSEQ + nwb + l15) * DMODEL;

    // number of live 4-tile (64-key) groups for this wave: hoisted, scalar
    int ngroups = (nwb <= qlast) ? ((qlast - nwb) >> 6) + 1 : 0;
    switch (ngroups) {
        case 1: score_loop<1>(sacc, qrow, kbase, klo, kb); break;
        case 2: score_loop<2>(sacc, qrow, kbase, klo, kb); break;
        case 3: score_loop<3>(sacc, qrow, kbase, klo, kb); break;
        case 4: score_loop<4>(sacc, qrow, kbase, klo, kb); break;
        default: break;
    }

    // mask + scale + per-row max (rows r / r+8 per lane half)
    float mr[8];
#pragma unroll
    for (int r = 0; r < 8; ++r) mr[r] = -3.0e38f;
#pragma unroll
    for (int tn = 0; tn < 16; ++tn) {
        if (nwb + tn * 16 > qlast) continue;
        int col = nwb + tn * 16 + l15;
#pragma unroll
        for (int r = 0; r < 8; ++r) {
            int rowq = q0 + r + hi8;
            float v = sacc[tn][r] * scale;
            if (col > rowq) v = -3.0e38f;
            sacc[tn][r] = v;
            mr[r] = fmaxf(mr[r], v);
        }
    }
#pragma unroll
    for (int r = 0; r < 8; ++r) {
        float m = mr[r];
#pragma unroll
        for (int off = 1; off < 16; off <<= 1)
            m = fmaxf(m, __shfl_xor(m, off, 32));
        if (l15 == 0) red[wave * 16 + r + hi8] = m;
    }
    __syncthreads();
    if (tid < 16) {
        float m = -3.0e38f;
#pragma unroll
        for (int w = 0; w < 8; ++w) m = fmaxf(m, red[w * 16 + tid]);
        rsta[tid] = m;
    }
    __syncthreads();

    // exp, row-sum, write unnormalized P (bf16) to LDS
    float rm[8];
#pragma unroll
    for (int r = 0; r < 8; ++r) rm[r] = rsta[r + hi8];
    float sr[8];
#pragma unroll
    for (int r = 0; r < 8; ++r) sr[r] = 0.0f;
#pragma unroll
    for (int tn = 0; tn < 16; ++tn) {
        if (nwb + tn * 16 > qlast) continue;
        int col = nwb + tn * 16 + l15;
#pragma unroll
        for (int r = 0; r < 8; ++r) {
            float p = __expf(sacc[tn][r] - rm[r]);
            sr[r] += p;
            sP[(r + hi8) * NSEQ + col] = (__bf16)p;
        }
    }
#pragma unroll
    for (int r = 0; r < 8; ++r) {
        float s = sr[r];
#pragma unroll
        for (int off = 1; off < 16; off <<= 1)
            s += __shfl_xor(s, off, 32);
        if (l15 == 0) red[wave * 16 + r + hi8] = s;
    }
    __syncthreads();
    if (tid < 16) {
        float s = 0.0f;
#pragma unroll
        for (int w = 0; w < 8; ++w) s += red[w * 16 + tid];
        rsta[16 + tid] = 1.0f / s;
    }
    __syncthreads();

    // ---------------- O = P @ V ----------------
    const int dwb = wave * 128;  // scalar
    v8f oacc[8];
#pragma unroll
    for (int t = 0; t < 8; ++t) oacc[t] = zero8();

    const __bf16* vbase = Vt + ((size_t)b * DMODEL + dwb + l15) * NSEQ;
    const int nkeys = q0 + 16;
    for (int n0 = 0; n0 < nkeys; n0 += 32) {
        v16bf a  = load_a16(sP + l15 * NSEQ + n0 + klo);
        v16bf nb = load_b16(vbase + n0 + klo);
#pragma unroll
        for (int t = 0; t < 8; ++t) {
            v16bf cb = nb;
            if (t + 1 < 8)
                nb = load_b16(vbase + (size_t)((t + 1) * 16) * NSEQ + n0 + klo);
            oacc[t] = wmma_bf16(a, cb, oacc[t]);
        }
    }

    // normalize -> stage O in LDS (sP dead now) -> coalesced b128 stores
    __syncthreads();
#pragma unroll
    for (int r = 0; r < 8; ++r) {
        int rloc = r + hi8;
        float inv = rsta[16 + rloc];
#pragma unroll
        for (int td = 0; td < 8; ++td)
            sO[rloc * DMODEL + dwb + td * 16 + l15] = oacc[td][r] * inv;
    }
    __syncthreads();
#pragma unroll
    for (int i = 0; i < 16; ++i) {
        int idx = tid + i * 256;            // 4096 uint4 total
        int row = idx >> 8, q = idx & 255;  // 256 uint4 per 1024-f32 row
        *(uint4*)(out + ((size_t)b * NSEQ + q0 + row) * DMODEL + q * 4) =
            ((const uint4*)sO)[idx];
    }
}

// ---------------------------------------------------------------------------
extern "C" void kernel_launch(void* const* d_in, const int* in_sizes, int n_in,
                              void* d_out, int out_size, void* d_ws, size_t ws_size,
                              hipStream_t stream) {
    const float* x  = (const float*)d_in[0];
    const float* Wk = (const float*)d_in[1];
    const float* Wq = (const float*)d_in[2];
    const float* Wv = (const float*)d_in[3];
    float* out = (float*)d_out;

    // workspace: Qb[8192][1024] bf16, Kb same, Vt[4][1024][2048] bf16 (48 MB)
    __bf16* Qb = (__bf16*)d_ws;
    __bf16* Kb = Qb + (size_t)MTOT * DMODEL;
    __bf16* Vt = Kb + (size_t)MTOT * DMODEL;

    dim3 blk(256);
    dim3 g1(MTOT / 128, DMODEL / 128);
    hipLaunchKernelGGL((qkv_gemm<0>), g1, blk, 0, stream, x, Wq, Qb);
    hipLaunchKernelGGL((qkv_gemm<0>), g1, blk, 0, stream, x, Wk, Kb);
    hipLaunchKernelGGL((qkv_gemm<1>), g1, blk, 0, stream, x, Wv, Vt);

    dim3 g2(NSEQ / 16, NBATCH);
    size_t shm = (size_t)16 * NSEQ * 2 + (8 * 16 + 32) * sizeof(float);
    hipLaunchKernelGGL(attn_kernel, g2, blk, shm, stream, Qb, Kb, Vt, out,
                       0.03125f /* 1/sqrt(1024) */);
}
